// MSHGNN_26096221290896
// MI455X (gfx1250) — matrix operand order repaired and checked by reading, not compiled, not run on hardware
//
#include <hip/hip_runtime.h>
#include <hip/hip_bf16.h>

#define D 128

typedef __attribute__((ext_vector_type(16))) __bf16 v16bf;
typedef __attribute__((ext_vector_type(8)))  float  v8f;
typedef __attribute__((ext_vector_type(2)))  __bf16 bf16x2;

// ---------- helpers ----------
__device__ __forceinline__ unsigned short f2bf(float f) {
    unsigned u = __builtin_bit_cast(unsigned, f);
    u += 0x7FFFu + ((u >> 16) & 1u);          // round-to-nearest-even
    return (unsigned short)(u >> 16);
}
__device__ __forceinline__ unsigned encf(float f) {
    unsigned u = __builtin_bit_cast(unsigned, f);
    return (u & 0x80000000u) ? ~u : (u | 0x80000000u);   // order-preserving key
}
__device__ __forceinline__ float decf(unsigned k) {
    unsigned u = (k & 0x80000000u) ? (k & 0x7FFFFFFFu) : ~k;
    return __builtin_bit_cast(float, u);
}

// ---------- conversion kernels ----------
__global__ void cvt_f32_bf16(const float* __restrict__ x,
                             unsigned short* __restrict__ y, int n) {
    int i = blockIdx.x * blockDim.x + threadIdx.x;
    if (i < n) y[i] = f2bf(x[i]);
}

// Pack all 8 W matrices into WMMA B-fragment layout:
//   frag base = ((r*4+kt)*8+nt)*256 ; per lane 8 dwords (bf16 pairs along K)
//   B order used here (contiguous k, same as A): lane<16 -> col=lane, K=0..15 ; lane>=16 -> col=lane-16, K=16..31
__global__ void build_wfrag(const float* __restrict__ W, unsigned int* __restrict__ wf) {
    int t = blockIdx.x * blockDim.x + threadIdx.x;   // 65536 threads
    if (t >= 8 * 4 * 8 * 32 * 8) return;
    int v    = t & 7;
    int lane = (t >> 3) & 31;
    int nt   = (t >> 8) & 7;
    int kt   = (t >> 11) & 3;
    int r    = (t >> 13) & 7;
    int n = nt * 16 + (lane & 15);
    int k = kt * 32 + (lane & 16) + 2 * v;
    const float* Wr = W + r * D * D;
    unsigned lo = f2bf(Wr[k * D + n]);
    unsigned hi = f2bf(Wr[(k + 1) * D + n]);
    wf[t] = lo | (hi << 16);
}

// wl[r] = W[r] @ attn_l[r] ; wr[r] = W[r] @ attn_r[r]   (128-vectors)
__global__ void calc_wlr(const float* __restrict__ W, const float* __restrict__ al,
                         const float* __restrict__ ar, float* __restrict__ wl8,
                         float* __restrict__ wr8) {
    int r = blockIdx.x, k = threadIdx.x;
    const float* Wr  = W + r * D * D + k * D;
    const float* alr = al + r * D;
    const float* arr = ar + r * D;
    float sl = 0.f, sr = 0.f;
    for (int j = 0; j < D; ++j) { float w = Wr[j]; sl += w * alr[j]; sr += w * arr[j]; }
    wl8[r * D + k] = sl;
    wr8[r * D + k] = sr;
}

// out = 2*(sum lam)*feat_dst + sum_g lam_g*(bias[2g]+bias[2g+1])
__global__ void init_out(const float* __restrict__ xd, const float* __restrict__ bias,
                         const float* __restrict__ lam, float* __restrict__ out, int total) {
    int i = blockIdx.x * blockDim.x + threadIdx.x;
    if (i >= total) return;
    int dcol = i & (D - 1);
    float ls = lam[0] + lam[1] + lam[2] + lam[3];
    float b = 0.f;
    #pragma unroll
    for (int g = 0; g < 4; ++g)
        b += lam[g] * (bias[(2 * g) * D + dcol] + bias[(2 * g + 1) * D + dcol]);
    out[i] = 2.f * ls * xd[i] + b;
}

// ---------- WMMA GEMM: hs[N,128] = xs_bf16[N,128] @ W_r(bf16 frag) ----------
// One wave -> 16x128 output strip. Per k-step: preload A frag + all 8 B frags
// (one load clause), then 8 back-to-back v_wmma_f32_16x16x32_bf16.
__global__ __launch_bounds__(128) void gemm_wmma(
        const unsigned short* __restrict__ Abf, const unsigned int* __restrict__ Wf,
        float* __restrict__ C, int nrows, int rbase) {
    int lane = threadIdx.x & 31;
    int wave = threadIdx.x >> 5;
    int row0 = (blockIdx.x * 4 + wave) * 16;

    v8f acc[8] = {};   // 8 column tiles of 16x16 f32

    int m = lane & 15;
    int arow = row0 + m;
    if (arow >= nrows) arow = nrows - 1;               // clamp tail loads
    const unsigned int* Arow = (const unsigned int*)(Abf + (size_t)arow * D);
    int khalf = (lane & 16) ? 8 : 0;

    for (int kt = 0; kt < 4; ++kt) {
        // A fragment: 16-bit A 16x32 layout
        v16bf a;
        #pragma unroll
        for (int v = 0; v < 8; ++v) {
            int k = kt * 32 + 2 * khalf + v * 2;     // even
            bf16x2 p = __builtin_bit_cast(bf16x2, Arow[k >> 1]);
            a[2 * v] = p[0];
            a[2 * v + 1] = p[1];
        }
        // Preload ALL eight B fragments for this k-step (decouples loads from WMMAs)
        v16bf b[8];
        #pragma unroll
        for (int nt = 0; nt < 8; ++nt) {
            const unsigned int* bp = Wf + rbase + (kt * 8 + nt) * 256 + lane * 8;
            #pragma unroll
            for (int v = 0; v < 8; ++v) {
                bf16x2 p = __builtin_bit_cast(bf16x2, bp[v]);
                b[nt][2 * v] = p[0];
                b[nt][2 * v + 1] = p[1];
            }
        }
        // Back-to-back WMMA chain
        #pragma unroll
        for (int nt = 0; nt < 8; ++nt) {
            acc[nt] = __builtin_amdgcn_wmma_f32_16x16x32_bf16(
                false, a, false, b[nt], (short)0, acc[nt], false, false);
        }
    }
    // store: 16x16 f32 C/D layout -> lane holds col n, rows j(+8)
    int n = lane & 15;
    int mbase = (lane & 16) ? 8 : 0;
    #pragma unroll
    for (int j = 0; j < 8; ++j) {
        int row = row0 + mbase + j;
        if (row < nrows) {
            float* cp = C + (size_t)row * D + n;
            #pragma unroll
            for (int nt = 0; nt < 8; ++nt) cp[nt * 16] = acc[nt][j];
        }
    }
}

// ---------- attention logits: el = xs@wl, er = xd@wr ----------
__global__ __launch_bounds__(256) void calc_eler(
        const float* __restrict__ xs, const float* __restrict__ xd,
        const float* __restrict__ wl, const float* __restrict__ wr,
        float* __restrict__ el, float* __restrict__ er, int n) {
    int i = blockIdx.x * blockDim.x + threadIdx.x;
    if (i >= n) return;
    const float4* xsp = (const float4*)(xs + (size_t)i * D);
    const float4* xdp = (const float4*)(xd + (size_t)i * D);
    const float4* wlp = (const float4*)wl;
    const float4* wrp = (const float4*)wr;
    float sl = 0.f, sr = 0.f;
    #pragma unroll 8
    for (int j = 0; j < D / 4; ++j) {
        float4 a = xsp[j], w = wlp[j];
        sl += a.x * w.x + a.y * w.y + a.z * w.z + a.w * w.w;
        float4 c = xdp[j], u = wrp[j];
        sr += c.x * u.x + c.y * u.y + c.z * u.z + c.w * u.w;
    }
    el[i] = sl;
    er[i] = sr;
}

__global__ void init_seg(unsigned* __restrict__ mkey, float* __restrict__ den, int n) {
    int i = blockIdx.x * blockDim.x + threadIdx.x;
    if (i < n) { mkey[i] = 0u; den[i] = 0.f; }
}

// pass 1: e = leaky_relu(el[s]+er[d]); segment max via ordered-uint atomicMax
__global__ void edge_pass1(const int* __restrict__ s, const int* __restrict__ d,
                           const float* __restrict__ el, const float* __restrict__ er,
                           float* __restrict__ ebuf, unsigned* __restrict__ mkey, int ne) {
    int i = blockIdx.x * blockDim.x + threadIdx.x;
    if (i >= ne) return;
    float e = el[s[i]] + er[d[i]];
    e = e > 0.f ? e : 0.2f * e;
    ebuf[i] = e;
    atomicMax(&mkey[d[i]], encf(e));
}

// pass 2: ex = exp(e - m[d]); den[d] += ex
__global__ void edge_pass2(const int* __restrict__ d, float* __restrict__ ebuf,
                           const unsigned* __restrict__ mkey, float* __restrict__ den, int ne) {
    int i = blockIdx.x * blockDim.x + threadIdx.x;
    if (i >= ne) return;
    int di = d[i];
    unsigned k = mkey[di];
    float m = (k == 0u) ? 0.f : decf(k);
    float ex = __expf(ebuf[i] - m);
    ebuf[i] = ex;
    atomicAdd(&den[di], ex);
}

// pass 3: out[d] += lam_g * (ex/den[d]) * hs[s]   (one wave per edge, float4 lanes)
__global__ __launch_bounds__(256) void edge_agg(
        const int* __restrict__ s, const int* __restrict__ d,
        const float* __restrict__ ebuf, const float* __restrict__ den,
        const float* __restrict__ hs, float* __restrict__ out,
        const float* __restrict__ lam, int g, int ne) {
    int wid  = (blockIdx.x * blockDim.x + threadIdx.x) >> 5;
    int lane = threadIdx.x & 31;
    if (wid >= ne) return;
    int si = s[wid], di = d[wid];
    float sc = lam[g] * ebuf[wid] / fmaxf(den[di], 1e-9f);
    float4 hv = ((const float4*)(hs + (size_t)si * D))[lane];
    float* op = out + (size_t)di * D + lane * 4;
    atomicAdd(op + 0, sc * hv.x);
    atomicAdd(op + 1, sc * hv.y);
    atomicAdd(op + 2, sc * hv.z);
    atomicAdd(op + 3, sc * hv.w);
}

// ---------- host ----------
extern "C" void kernel_launch(void* const* d_in, const int* in_sizes, int n_in,
                              void* d_out, int out_size, void* d_ws, size_t ws_size,
                              hipStream_t stream) {
    const float* feat_src = (const float*)d_in[0];
    const float* feat_dst = (const float*)d_in[1];
    const int* intra_src  = (const int*)d_in[2];
    const int* intra_dst  = (const int*)d_in[3];
    const int* inter_src  = (const int*)d_in[4];
    const int* inter_dst  = (const int*)d_in[5];
    const float* W      = (const float*)d_in[6];
    const float* attn_l = (const float*)d_in[7];
    const float* attn_r = (const float*)d_in[8];
    const float* bias   = (const float*)d_in[9];
    const float* lam    = (const float*)d_in[10];
    float* out = (float*)d_out;

    const int N = in_sizes[0] / D;
    const int E = in_sizes[2];

    // workspace carve-out
    auto au = [](size_t x) { return (x + 255) & ~(size_t)255; };
    char* p = (char*)d_ws;
    size_t off = 0;
    unsigned short* fsrc_bf = (unsigned short*)(p + off); off += au((size_t)N * D * 2);
    unsigned short* fdst_bf = (unsigned short*)(p + off); off += au((size_t)N * D * 2);
    unsigned int*   wfrag   = (unsigned int*)(p + off);   off += au((size_t)65536 * 4);
    float* wl8  = (float*)(p + off); off += au((size_t)8 * D * 4);
    float* wr8  = (float*)(p + off); off += au((size_t)8 * D * 4);
    float* el   = (float*)(p + off); off += au((size_t)N * 4);
    float* er   = (float*)(p + off); off += au((size_t)N * 4);
    unsigned* mkey = (unsigned*)(p + off); off += au((size_t)N * 4);
    float* den  = (float*)(p + off); off += au((size_t)N * 4);
    float* ebuf = (float*)(p + off); off += au((size_t)E * 4);
    float* hs   = (float*)(p + off); off += au((size_t)N * D * 4);

    const int total = N * D;
    cvt_f32_bf16<<<(total + 255) / 256, 256, 0, stream>>>(feat_src, fsrc_bf, total);
    cvt_f32_bf16<<<(total + 255) / 256, 256, 0, stream>>>(feat_dst, fdst_bf, total);
    build_wfrag<<<256, 256, 0, stream>>>(W, wfrag);
    calc_wlr<<<8, 128, 0, stream>>>(W, attn_l, attn_r, wl8, wr8);
    init_out<<<(total + 255) / 256, 256, 0, stream>>>(feat_dst, bias, lam, out, total);

    for (int r = 0; r < 8; ++r) {
        int g = r >> 1;
        bool use_src = (r == 0 || r == 1 || r == 4 || r == 5);   // conv1/conv3 take feat_src
        const float* xs_f = use_src ? feat_src : feat_dst;
        const unsigned short* xs_bf = use_src ? fsrc_bf : fdst_bf;
        bool intra = ((r & 1) == 0);
        const int* a = intra ? intra_src : inter_src;
        const int* b = intra ? intra_dst : inter_dst;
        const int* s_idx = (r < 4) ? a : b;   // conv3/conv4 use reversed graph
        const int* d_idx = (r < 4) ? b : a;

        gemm_wmma<<<(N + 63) / 64, 128, 0, stream>>>(xs_bf, wfrag, hs, N, r * 4 * 8 * 256);
        calc_eler<<<(N + 255) / 256, 256, 0, stream>>>(xs_f, feat_dst, wl8 + r * D,
                                                       wr8 + r * D, el, er, N);
        init_seg<<<(N + 255) / 256, 256, 0, stream>>>(mkey, den, N);
        edge_pass1<<<(E + 255) / 256, 256, 0, stream>>>(s_idx, d_idx, el, er, ebuf, mkey, E);
        edge_pass2<<<(E + 255) / 256, 256, 0, stream>>>(d_idx, ebuf, mkey, den, E);
        edge_agg<<<(E * 32 + 255) / 256, 256, 0, stream>>>(s_idx, d_idx, ebuf, den, hs,
                                                           out, lam, g, E);
    }
    (void)n_in; (void)out_size; (void)ws_size;
}
